// CRFB_54065048322579
// MI455X (gfx1250) — compile-verified, matrix operand order checked by reading, not run on hardware
//
#include <hip/hip_runtime.h>
#include <hip/hip_bf16.h>

#define BB 8
#define CC 144
#define HH 192
#define WW 192
#define HWP (HH*WW)            // 36864
#define NHEADS 6
#define CHD 24                 // C / HEADS
#define HID 288
#define CH2 576                // 2*HID

typedef __attribute__((ext_vector_type(16))) _Float16 v16h;
typedef __attribute__((ext_vector_type(8)))  _Float16 v8h;
typedef __attribute__((ext_vector_type(8)))  float    v8f;
typedef __attribute__((ext_vector_type(4)))  unsigned int u32x4;
typedef __attribute__((ext_vector_type(8)))  int i32x8;
typedef __attribute__((ext_vector_type(4)))  int i32x4;

// TDM builtins are device-only: probe them ONLY on the device pass.
// Host pass compiles the (never-executed-on-device) fallback so parsing works.
#if defined(__HIP_DEVICE_COMPILE__)
#  if !(__has_builtin(__builtin_amdgcn_tensor_load_to_lds) && __has_builtin(__builtin_amdgcn_s_wait_tensorcnt))
#    error "gfx1250 Tensor Data Mover builtins not available on device toolchain"
#  endif
#  define HAVE_TDM 1
#else
#  define HAVE_TDM 0
#endif

#if HAVE_TDM
// 2D tile DMA: global (row-major, stride0 elements of 2 bytes) -> contiguous LDS.
// D# layout per CDNA5 ISA ch.8: group0 = {count, lds_addr, global_addr, type=2},
// group1 = {data_size=2B, tensor dims, tile dims, dim0 stride}.
__device__ __forceinline__ void tdm_load_2d_f16(const _Float16* gsrc, unsigned lds_byte_off,
                                                unsigned tile_d0, unsigned tile_d1,
                                                unsigned tensor_d0, unsigned tensor_d1,
                                                unsigned long long stride0_elems) {
    unsigned long long ga = (unsigned long long)(uintptr_t)gsrc;
    u32x4 g0;
    g0[0] = 1u;                                              // count=1 (valid user D#)
    g0[1] = lds_byte_off;                                    // lds_addr
    g0[2] = (unsigned)(ga & 0xffffffffu);                    // global_addr[31:0]
    g0[3] = (unsigned)((ga >> 32) & 0x01ffffffu) | (2u << 30); // addr[56:32] | type=2
    i32x8 g1;
    g1[0] = (int)(1u << 16);                                 // data_size=1 -> 2 bytes
    g1[1] = (int)((tensor_d0 & 0xffffu) << 16);              // tensor_dim0[15:0]
    g1[2] = (int)(((tensor_d0 >> 16) & 0xffffu) | ((tensor_d1 & 0xffffu) << 16));
    g1[3] = (int)(((tensor_d1 >> 16) & 0xffffu) | ((tile_d0 & 0xffffu) << 16));
    g1[4] = (int)(tile_d1 & 0xffffu);                        // tile_dim1 (tile_dim2=0)
    g1[5] = (int)(unsigned)(stride0_elems & 0xffffffffu);    // dim0_stride[31:0]
    g1[6] = (int)(unsigned)((stride0_elems >> 32) & 0xffffu);
    g1[7] = 0;
    i32x4 gz = {0, 0, 0, 0};
#if __clang_major__ >= 23
    i32x8 gz8 = {0, 0, 0, 0, 0, 0, 0, 0};
    __builtin_amdgcn_tensor_load_to_lds(g0, g1, gz, gz, gz8, 0);
#else
    __builtin_amdgcn_tensor_load_to_lds(g0, g1, gz, gz, 0);
#endif
}
#endif

// ---------------- zero small accumulators ----------------
__global__ void k_zero(float* __restrict__ p, int n) {
    int i = blockIdx.x * blockDim.x + threadIdx.x;
    if (i < n) p[i] = 0.f;
}

// ---------------- weight f32 -> f16 conversion (tiny, once per launch) ----------------
__global__ void k_wcvt(const float* __restrict__ src, _Float16* __restrict__ dst, int n) {
    int i = blockIdx.x * blockDim.x + threadIdx.x;
    if (i < n) dst[i] = (_Float16)src[i];
}

// ---------------- channel LayerNorm: f32 -> f16, single pass ----------------
// Channel values cached in VGPRs (144 f32; CDNA5 waves have up to 1024 VGPRs).
__global__ __launch_bounds__(256)
void k_ln(const float* __restrict__ x, const float* __restrict__ w,
          const float* __restrict__ b, _Float16* __restrict__ y) {
    int pg = blockIdx.x * blockDim.x + threadIdx.x;    // over B*HW
    if (pg >= BB * HWP) return;
    int bb = pg / HWP, p = pg - bb * HWP;
    const float* xp = x + (size_t)bb * CC * HWP + p;
    float v[CC];
    float s = 0.f, ss = 0.f;
#pragma unroll
    for (int c = 0; c < CC; ++c) {
        v[c] = xp[(size_t)c * HWP];
        s += v[c]; ss += v[c] * v[c];
    }
    float mu  = s * (1.f / CC);
    float var = ss * (1.f / CC) - mu * mu;
    float rs  = rsqrtf(var + 1e-5f);
    _Float16* yp = y + (size_t)bb * CC * HWP + p;
#pragma unroll
    for (int c = 0; c < CC; ++c)
        yp[(size_t)c * HWP] = (_Float16)((v[c] - mu) * rs * w[c] + b[c]);
}

// ---------------- 1x1 conv as WMMA GEMM ----------------
// Out[b, m, n] = sum_k Wh[m,k] * X[b,k,n]   (both f16, f32 accumulate)
// Block: 256 thr = 8 waves; tile 48(M) x 128(N); wave -> 48x16 strip (3 wmma / K-step).
// Weight slab [48][Cin] DMA'd to LDS by the Tensor Data Mover (wave 0, TENSORcnt).
// X tile double-buffered in LDS; zero-padded K tail on the B side makes the
// unpadded A slab safe. Grid: x = M-blocks (fastest) so consecutive workgroups
// share the same X tile -> X re-reads stay in L2/WGP$.
// EPI 0: store f16.  EPI 1: store f32 = acc + Res (residual fusion).
template <int EPI>
__global__ __launch_bounds__(256)
void k_gemm(const _Float16* __restrict__ Wh, const _Float16* __restrict__ X,
            void* __restrict__ Out, const float* __restrict__ Res,
            int Cout, int Cin) {
    extern __shared__ _Float16 lds[];
    const int KPAD = (Cin + 31) & ~31;
    _Float16* Wl  = lds;                       // [48][Cin] + 64 slack
    _Float16* Xl0 = lds + 48 * Cin + 64;       // [2][128][32]
    const unsigned XBUF = 128 * 32;            // halves per X buffer

    const int tid = threadIdx.x;
    const int m0  = blockIdx.x * 48;
    const int n0  = blockIdx.y * 128;
    const int b   = blockIdx.z;

    // zero the slack region (protects row-47 tail fragment reads)
    if (tid < 64) Wl[48 * Cin + tid] = (_Float16)0.f;

#if HAVE_TDM
    if (tid == 0) {
        tdm_load_2d_f16(Wh + (size_t)m0 * Cin, (unsigned)(uintptr_t)Wl,
                        (unsigned)Cin, 48u, (unsigned)Cin, 48u,
                        (unsigned long long)Cin);
        __builtin_amdgcn_s_wait_tensorcnt(0);
    }
#else
    for (int idx = tid; idx < 48 * Cin; idx += 256)
        Wl[idx] = Wh[(size_t)m0 * Cin + idx];
#endif

    const int ns = tid & 127;          // staging column within tile
    const int kb = (tid >> 7) * 16;    // staging k sub-block {0,16}; wave-uniform
    const _Float16* Xcol = X + (size_t)b * Cin * HWP + n0 + ns;

    v8h t0, t1;
    auto fetch = [&](int kk2, v8h& va, v8h& vb) {
        const int kbase = kk2 + kb;            // multiple of 16; Cin%16==0
        if (kbase < Cin) {                     // wave-uniform -> scalar branch
#pragma unroll
            for (int j = 0; j < 8; ++j) va[j] = Xcol[(size_t)(kbase + j) * HWP];
#pragma unroll
            for (int j = 0; j < 8; ++j) vb[j] = Xcol[(size_t)(kbase + 8 + j) * HWP];
        } else {
#pragma unroll
            for (int j = 0; j < 8; ++j) { va[j] = (_Float16)0.f; vb[j] = (_Float16)0.f; }
        }
    };

    // prologue: tile 0 -> buffer 0
    fetch(0, t0, t1);
    *(v8h*)&Xl0[ns * 32 + kb]     = t0;
    *(v8h*)&Xl0[ns * 32 + kb + 8] = t1;

    v8f acc[3] = {};
    const int wave  = tid >> 5, lane = tid & 31;
    const int lhalf = lane & 15;
    const bool hi   = lane >= 16;
    const int o1 = hi ? 8 : 0, o2 = hi ? 24 : 16;
    const int nfr = wave * 16 + lhalf;         // fragment column
    unsigned xoff = 0;                         // current buffer offset (halves)

    for (int kk = 0; kk < KPAD; kk += 32) {
        __syncthreads();                        // current buffer ready
        const bool last = (kk + 32 >= KPAD);
        if (!last) fetch(kk + 32, t0, t1);      // global loads overlap math below
        _Float16* cur = Xl0 + xoff;
        _Float16* nxt = Xl0 + (xoff ^ XBUF);
        xoff ^= XBUF;

        // B fragment: 32(K) x 16(N); lane<16: K{0..7,16..23}, lane>=16: K{8..15,24..31}
        v8h blo = *(v8h*)&cur[nfr * 32 + o1];
        v8h bhi = *(v8h*)&cur[nfr * 32 + o2];
        v16h bfrag;
#pragma unroll
        for (int i = 0; i < 8; ++i) { bfrag[i] = blo[i]; bfrag[8 + i] = bhi[i]; }

#pragma unroll
        for (int mt = 0; mt < 3; ++mt) {
            const int m = mt * 16 + lhalf;
            v8h alo = *(v8h*)&Wl[m * Cin + kk + o1];
            v8h ahi = *(v8h*)&Wl[m * Cin + kk + o2];
            v16h afrag;
#pragma unroll
            for (int i = 0; i < 8; ++i) { afrag[i] = alo[i]; afrag[8 + i] = ahi[i]; }
            acc[mt] = __builtin_amdgcn_wmma_f32_16x16x32_f16(
                false, afrag, false, bfrag, (short)0, acc[mt], false, false);
        }

        if (!last) {                            // write next tile to alternate buffer
            *(v8h*)&nxt[ns * 32 + kb]     = t0;
            *(v8h*)&nxt[ns * 32 + kb + 8] = t1;
        }
    }

    // epilogue: C/D layout -> lane<16: M=r, N=lane ; lane>=16: M=8+r, N=lane-16
    const int ncol = n0 + wave * 16 + lhalf;
#pragma unroll
    for (int mt = 0; mt < 3; ++mt) {
#pragma unroll
        for (int r = 0; r < 8; ++r) {
            int m = m0 + mt * 16 + (hi ? 8 + r : r);
            size_t oidx = ((size_t)b * Cout + m) * HWP + ncol;
            float v = acc[mt][r];
            if (EPI == 0) ((_Float16*)Out)[oidx] = (_Float16)v;
            else          ((float*)Out)[oidx]    = v + Res[oidx];
        }
    }
}

// ---------------- depthwise 3x3 (SAME) + optional L2-norm accumulation ----------------
__global__ __launch_bounds__(192)
void k_dw3(const _Float16* __restrict__ in, const float* __restrict__ w,
           _Float16* __restrict__ out, float* __restrict__ norm,
           int Cg, int norm_limit) {
    const int h = blockIdx.x, wd = threadIdx.x;
    const int c = blockIdx.y, b = blockIdx.z;
    const _Float16* ip = in + ((size_t)b * Cg + c) * HWP;
    const float* wc = w + c * 9;
    const int wm = (wd == 0) ? 0 : wd - 1;
    const int wp = (wd == WW - 1) ? WW - 1 : wd + 1;
    float acc = 0.f;
#pragma unroll
    for (int dy = -1; dy <= 1; ++dy) {
        int hh = h + dy;
        if (hh < 0 || hh >= HH) continue;      // uniform
        const _Float16* row = ip + hh * WW;
        float xm = (float)row[wm];
        float x0 = (float)row[wd];
        float xp = (float)row[wp];
        xm = (wd == 0) ? 0.f : xm;
        xp = (wd == WW - 1) ? 0.f : xp;
        acc += wc[(dy + 1) * 3 + 0] * xm + wc[(dy + 1) * 3 + 1] * x0
             + wc[(dy + 1) * 3 + 2] * xp;
    }
    out[((size_t)b * Cg + c) * HWP + h * WW + wd] = (_Float16)acc;

    __shared__ float red[256];
    if (threadIdx.x < 64) red[192 + threadIdx.x] = 0.f;
    red[threadIdx.x] = (c < norm_limit) ? acc * acc : 0.f;
    __syncthreads();
    for (int s = 128; s > 0; s >>= 1) {
        if (threadIdx.x < s) red[threadIdx.x] += red[threadIdx.x + s];
        __syncthreads();
    }
    if (threadIdx.x == 0 && c < norm_limit)
        atomicAdd(&norm[(size_t)b * norm_limit + c], red[0]);
}

// ---------------- channel attention: S[c,d] += qn[c,i]*kn[d,i] ----------------
#define QK_CHUNKS 32
#define TI 64
__global__ __launch_bounds__(256)
void k_attn_qk(const _Float16* __restrict__ q, const _Float16* __restrict__ kvd,
               const float* __restrict__ normq, const float* __restrict__ normk,
               float* __restrict__ S) {
    const int chunk = blockIdx.x, head = blockIdx.y, b = blockIdx.z;
    const int tid = threadIdx.x;
    __shared__ float sq[CHD], sk[CHD];
    __shared__ float qs[CHD][TI], ks[CHD][TI];
    if (tid < CHD)
        sq[tid] = 1.f / fmaxf(sqrtf(normq[b * CC + head * CHD + tid]), 1e-12f);
    else if (tid < 2 * CHD)
        sk[tid - CHD] = 1.f / fmaxf(sqrtf(normk[b * CC + head * CHD + (tid - CHD)]), 1e-12f);

    const int i0 = chunk * (HWP / QK_CHUNKS);
    const int i1 = i0 + (HWP / QK_CHUNKS);
    const int p0 = tid,       c0 = p0 / CHD, d0 = p0 % CHD;
    const int p1 = tid + 256, c1 = p1 / CHD, d1 = p1 % CHD;
    const int p2 = tid + 512;
    const bool has2 = p2 < CHD * CHD;
    const int c2 = has2 ? p2 / CHD : 0, d2 = has2 ? p2 % CHD : 0;
    float acc0 = 0.f, acc1 = 0.f, acc2 = 0.f;

    const _Float16* qb = q   + ((size_t)b * CC + head * CHD) * HWP;
    const _Float16* kb = kvd + ((size_t)b * 2 * CC + head * CHD) * HWP;
    for (int i = i0; i < i1; i += TI) {
        __syncthreads();
        for (int idx = tid; idx < CHD * TI; idx += 256) {
            int c = idx / TI, j = idx - c * TI;
            qs[c][j] = (float)qb[(size_t)c * HWP + i + j] * sq[c];
            ks[c][j] = (float)kb[(size_t)c * HWP + i + j] * sk[c];
        }
        __syncthreads();
#pragma unroll 8
        for (int j = 0; j < TI; ++j) {
            acc0 += qs[c0][j] * ks[d0][j];
            acc1 += qs[c1][j] * ks[d1][j];
            if (has2) acc2 += qs[c2][j] * ks[d2][j];
        }
    }
    float* Sb = S + ((size_t)b * NHEADS + head) * CHD * CHD;
    atomicAdd(&Sb[p0], acc0);
    atomicAdd(&Sb[p1], acc1);
    if (has2) atomicAdd(&Sb[p2], acc2);
}

// ---------------- softmax over 24-wide rows, scaled by temperature ----------------
__global__ void k_attn_softmax(const float* __restrict__ S, const float* __restrict__ temp,
                               float* __restrict__ P) {
    int bh = blockIdx.x;             // b*HEADS + h
    int head = bh % NHEADS;
    int r = threadIdx.x;
    if (r >= CHD) return;
    const float* Sr = S + (size_t)bh * CHD * CHD + r * CHD;
    float* Pr = P + (size_t)bh * CHD * CHD + r * CHD;
    float t = temp[head];
    float v[CHD];
    float m = -1e30f;
    for (int d = 0; d < CHD; ++d) { v[d] = Sr[d] * t; m = fmaxf(m, v[d]); }
    float s = 0.f;
    for (int d = 0; d < CHD; ++d) { v[d] = expf(v[d] - m); s += v[d]; }
    float inv = 1.f / s;
    for (int d = 0; d < CHD; ++d) Pr[d] = v[d] * inv;
}

// ---------------- out[c,n] = sum_d P[c,d] * v[d,n] ----------------
__global__ __launch_bounds__(256)
void k_attn_out(const float* __restrict__ P, const _Float16* __restrict__ kvd,
                _Float16* __restrict__ out) {
    const int head = blockIdx.y, b = blockIdx.z;
    const int i = blockIdx.x * 256 + threadIdx.x;
    __shared__ float Pl[CHD * CHD];
    for (int idx = threadIdx.x; idx < CHD * CHD; idx += 256)
        Pl[idx] = P[((size_t)b * NHEADS + head) * CHD * CHD + idx];
    __syncthreads();
    const _Float16* vb = kvd + ((size_t)b * 2 * CC + CC + head * CHD) * HWP + i;
    float vv[CHD];
#pragma unroll
    for (int d = 0; d < CHD; ++d) vv[d] = (float)vb[(size_t)d * HWP];
    _Float16* ob = out + ((size_t)b * CC + head * CHD) * HWP + i;
#pragma unroll
    for (int c = 0; c < CHD; ++c) {
        float o = 0.f;
#pragma unroll
        for (int d = 0; d < CHD; ++d) o += Pl[c * CHD + d] * vv[d];
        ob[(size_t)c * HWP] = (_Float16)o;
    }
}

// ---------------- FFN: dw3 on both halves + exact gelu gate ----------------
__global__ __launch_bounds__(192)
void k_ffn_dw_gelu(const _Float16* __restrict__ in, const float* __restrict__ w,
                   _Float16* __restrict__ gate) {
    const int h = blockIdx.x, wd = threadIdx.x;
    const int c = blockIdx.y, b = blockIdx.z;
    const _Float16* ip0 = in + ((size_t)b * CH2 + c) * HWP;
    const _Float16* ip1 = in + ((size_t)b * CH2 + c + HID) * HWP;
    const float* w0 = w + c * 9;
    const float* w1 = w + (c + HID) * 9;
    const int wm = (wd == 0) ? 0 : wd - 1;
    const int wp = (wd == WW - 1) ? WW - 1 : wd + 1;
    float a0 = 0.f, a1 = 0.f;
#pragma unroll
    for (int dy = -1; dy <= 1; ++dy) {
        int hh = h + dy;
        if (hh < 0 || hh >= HH) continue;      // uniform
        const int ro = hh * WW;
        float m0 = (float)ip0[ro + wm], z0 = (float)ip0[ro + wd], p0 = (float)ip0[ro + wp];
        float m1 = (float)ip1[ro + wm], z1 = (float)ip1[ro + wd], p1 = (float)ip1[ro + wp];
        m0 = (wd == 0) ? 0.f : m0;  p0 = (wd == WW - 1) ? 0.f : p0;
        m1 = (wd == 0) ? 0.f : m1;  p1 = (wd == WW - 1) ? 0.f : p1;
        int kr = (dy + 1) * 3;
        a0 += w0[kr] * m0 + w0[kr + 1] * z0 + w0[kr + 2] * p0;
        a1 += w1[kr] * m1 + w1[kr + 1] * z1 + w1[kr + 2] * p1;
    }
    float g = 0.5f * a0 * (1.f + erff(a0 * 0.70710678118654752f));
    gate[((size_t)b * HID + c) * HWP + h * WW + wd] = (_Float16)(g * a1);
}

// ---------------- host orchestration ----------------
extern "C" void kernel_launch(void* const* d_in, const int* in_sizes, int n_in,
                              void* d_out, int out_size, void* d_ws, size_t ws_size,
                              hipStream_t stream) {
    const float* low     = (const float*)d_in[0];
    const float* high    = (const float*)d_in[1];
    const float* w_q     = (const float*)d_in[2];
    const float* w_q_dw  = (const float*)d_in[3];
    const float* w_kv    = (const float*)d_in[4];
    const float* w_kv_dw = (const float*)d_in[5];
    const float* w_proj  = (const float*)d_in[6];
    const float* temp    = (const float*)d_in[7];
    const float* ln1_w   = (const float*)d_in[8];
    const float* ln1_b   = (const float*)d_in[9];
    const float* ln2_w   = (const float*)d_in[10];
    const float* ln2_b   = (const float*)d_in[11];
    const float* w_pin   = (const float*)d_in[12];
    const float* w_dw    = (const float*)d_in[13];
    const float* w_pout  = (const float*)d_in[14];
    float* out = (float*)d_out;

    const size_t S1  = (size_t)BB * CC * HWP;   // elements of one [B,C,HW] plane
    const size_t S1B = S1 * 2;                  // bytes as f16
    char* ws = (char*)d_ws;
    _Float16* lnlow   = (_Float16*)(ws + 0 * S1B);
    _Float16* lnhigh  = (_Float16*)(ws + 1 * S1B);
    _Float16* kvpre   = (_Float16*)(ws + 2 * S1B);   // 2*S1B
    _Float16* pinout  = (_Float16*)(ws + 0 * S1B);   // reuse 4*S1B (ln/kvpre dead)
    _Float16* qpre    = (_Float16*)(ws + 4 * S1B);
    _Float16* attnout = qpre;                        // reuse after dw3(q)
    _Float16* qbuf    = (_Float16*)(ws + 5 * S1B);
    _Float16* kvd     = (_Float16*)(ws + 6 * S1B);   // 2*S1B
    float*    xbuf    = (float*)  (ws + 8 * S1B);    // f32 residual, 2*S1B bytes
    _Float16* xln     = (_Float16*)(ws + 10 * S1B);
    _Float16* gate    = (_Float16*)(ws + 11 * S1B);  // 2*S1B
    float*    normq   = (float*)  (ws + 13 * S1B);
    float*    normk   = normq + BB * CC;
    float*    Sbuf    = normk + BB * CC;
    float*    Pbuf    = Sbuf + (size_t)BB * NHEADS * CHD * CHD;
    // converted f16 weights (contiguous region after small f32 buffers)
    _Float16* whq     = (_Float16*)(Pbuf + (size_t)BB * NHEADS * CHD * CHD);
    _Float16* whkv    = whq   + (size_t)CC * CC;
    _Float16* whproj  = whkv  + (size_t)2 * CC * CC;
    _Float16* whpin   = whproj + (size_t)CC * CC;
    _Float16* whpout  = whpin + (size_t)CH2 * CC;

    auto smem = [](int Cin) {
        return (size_t)(48 * Cin + 64 + 2 * 128 * 32) * sizeof(_Float16);
    };

    // zero norm + S accumulators (contiguous)
    int nzero = 2 * BB * CC + BB * NHEADS * CHD * CHD;
    k_zero<<<(nzero + 255) / 256, 256, 0, stream>>>(normq, nzero);

    // convert GEMM weights to f16 (tiny)
    k_wcvt<<<(CC * CC + 255) / 256, 256, 0, stream>>>(w_q, whq, CC * CC);
    k_wcvt<<<(2 * CC * CC + 255) / 256, 256, 0, stream>>>(w_kv, whkv, 2 * CC * CC);
    k_wcvt<<<(CC * CC + 255) / 256, 256, 0, stream>>>(w_proj, whproj, CC * CC);
    k_wcvt<<<(CH2 * CC + 255) / 256, 256, 0, stream>>>(w_pin, whpin, CH2 * CC);
    k_wcvt<<<(CC * HID + 255) / 256, 256, 0, stream>>>(w_pout, whpout, CC * HID);

    const int lnblocks = (BB * HWP + 255) / 256;
    k_ln<<<lnblocks, 256, 0, stream>>>(low,  ln1_w, ln1_b, lnlow);
    k_ln<<<lnblocks, 256, 0, stream>>>(high, ln1_w, ln1_b, lnhigh);

    // q = dw3(w_q x ln(high)) ; kv = dw3(w_kv x ln(low))
    k_gemm<0><<<dim3(CC / 48, HWP / 128, BB), 256, smem(CC), stream>>>(
        whq, lnhigh, (void*)qpre, nullptr, CC, CC);
    k_gemm<0><<<dim3(2 * CC / 48, HWP / 128, BB), 256, smem(CC), stream>>>(
        whkv, lnlow, (void*)kvpre, nullptr, 2 * CC, CC);
    k_dw3<<<dim3(HH, CC, BB), 192, 0, stream>>>(qpre, w_q_dw, qbuf, normq, CC, CC);
    k_dw3<<<dim3(HH, 2 * CC, BB), 192, 0, stream>>>(kvpre, w_kv_dw, kvd, normk, 2 * CC, CC);

    // transposed channel attention
    k_attn_qk<<<dim3(QK_CHUNKS, NHEADS, BB), 256, 0, stream>>>(qbuf, kvd, normq, normk, Sbuf);
    k_attn_softmax<<<BB * NHEADS, 32, 0, stream>>>(Sbuf, temp, Pbuf);
    k_attn_out<<<dim3(HWP / 256, NHEADS, BB), 256, 0, stream>>>(Pbuf, kvd, attnout);

    // x = high + proj(attnout)   (residual fused into GEMM epilogue)
    k_gemm<1><<<dim3(CC / 48, HWP / 128, BB), 256, smem(CC), stream>>>(
        whproj, attnout, (void*)xbuf, high, CC, CC);

    // FFN branch
    k_ln<<<lnblocks, 256, 0, stream>>>(xbuf, ln2_w, ln2_b, xln);
    k_gemm<0><<<dim3(CH2 / 48, HWP / 128, BB), 256, smem(CC), stream>>>(
        whpin, xln, (void*)pinout, nullptr, CH2, CC);
    k_ffn_dw_gelu<<<dim3(HH, HID, BB), 192, 0, stream>>>(pinout, w_dw, gate);

    // out = x + pout(gate)   (residual fused)
    k_gemm<1><<<dim3(CC / 48, HWP / 128, BB), 256, smem(HID), stream>>>(
        whpout, gate, (void*)out, xbuf, CC, HID);
}
